// SelfAttentionWithPolarPE_30219389894973
// MI455X (gfx1250) — matrix operand order repaired and checked by reading, not compiled, not run on hardware
//
#include <hip/hip_runtime.h>
#include <hip/hip_bf16.h>

// ---------------------------------------------------------------------------
// SelfAttentionWithPolarPE on gfx1250 (MI455X), wave32 + WMMA bf16.
//
// Decomposition (avoids the naive 172-GFLOP per-pair K/V projection):
//   k_sp[b,h,i,j] = k_base[b,h,j] + pe[b,i,j] @ Wk_pe
//   q . k_pe      = pe[b,i,j] . qWk[b,h,i]          (qWk = q . Wk_pe)
//   sum_j a*v_pe  = (sum_j a[i,j]*pe[i,j]) @ Wv_pe  ("S" contraction)
// pe is recomputed twice (score pass + output pass) instead of storing 68 MB.
// WMMA operands are fed with vectorized (b128-class) loads, except the scalar guarded qWk B operand.
// ---------------------------------------------------------------------------

#define HEADS     8
#define DIM_HEAD  64
#define DIM       512
#define INNER     512
#define NFREQ     32
#define POS_DIM   130     // 2*(2*32+1)
#define LS        256
#define LSEQ      272
#define BATCH     2
#define LPAD      288     // key length padded to multiple of 32
#define PPAD      160     // pos-dim padded to multiple of 32
#define PHYS_SCALE 51.5f
#define PI_F       3.14159265358979323846f
#define ATTN_SCALE 0.125f // 64^-0.5

typedef __attribute__((ext_vector_type(16))) __bf16 v16bf;
typedef __attribute__((ext_vector_type(8)))  __bf16 v8bf;
typedef __attribute__((ext_vector_type(8)))  float  v8f;
typedef __attribute__((ext_vector_type(4)))  float  v4f;

__device__ inline v8f wmma_bf16(const v16bf& a, const v16bf& b, const v8f& c) {
  // v_wmma_f32_16x16x32_bf16  D = A(16x32) * B(32x16) + C
  return __builtin_amdgcn_wmma_f32_16x16x32_bf16(false, a, false, b, (short)0, c,
                                                 false, false);
}

// ---- vectorized operand loaders ------------------------------------------
// A fragment (16x32 MxK): lanes 0-15 hold K 0-7/16-23, lanes 16-31 K 8-15/24-31,
// row M = lane&15.  Source: row-major bf16, contiguous in K -> two 16B loads.
__device__ inline v16bf load_a_vec(const __bf16* tile, size_t stride, int lane) {
  const int m  = lane & 15;
  const int kh = (lane >> 4) << 3; // 0 or 8
  const __bf16* rp = tile + (size_t)m * stride;
  v8bf lo = *(const v8bf*)(rp + kh);
  v8bf hi = *(const v8bf*)(rp + 16 + kh);
  return __builtin_shufflevector(lo, hi, 0, 1, 2, 3, 4, 5, 6, 7,
                                 8, 9, 10, 11, 12, 13, 14, 15);
}

// Same, converting an f32 row-major source on the fly (four 16B loads + cvt).
__device__ inline v16bf load_a_f32_vec(const float* tile, size_t stride, int lane) {
  const int m  = lane & 15;
  const int kh = (lane >> 4) << 3;
  const float* rp = tile + (size_t)m * stride;
  v4f a0 = *(const v4f*)(rp + kh);
  v4f a1 = *(const v4f*)(rp + kh + 4);
  v4f a2 = *(const v4f*)(rp + 16 + kh);
  v4f a3 = *(const v4f*)(rp + 16 + kh + 4);
  v16bf a;
#pragma unroll
  for (int i = 0; i < 4; ++i) {
    a[i]      = (__bf16)a0[i];
    a[4 + i]  = (__bf16)a1[i];
    a[8 + i]  = (__bf16)a2[i];
    a[12 + i] = (__bf16)a3[i];
  }
  return a;
}

// B fragment (32x16 KxN): lanes 0-15 hold K 0-7/16-23, lanes 16-31 K 8-15/24-31, col N =
// lane&15.  Source holds B^T (contiguous in K per column) -> two 16B loads.
__device__ inline v16bf load_bT_vec(const __bf16* tile, size_t stride, int lane) {
  const int n  = lane & 15;
  const int kh = (lane >> 4) << 3; // 0 or 8
  const __bf16* cp = tile + (size_t)n * stride;
  v8bf lo = *(const v8bf*)(cp + kh);
  v8bf hi = *(const v8bf*)(cp + 16 + kh);
  return __builtin_shufflevector(lo, hi, 0, 1, 2, 3, 4, 5, 6, 7,
                                 8, 9, 10, 11, 12, 13, 14, 15);
}

// generic (scalar) B loader for guarded / strided sources
template <typename F>
__device__ inline v16bf load_b_gen(int lane, F&& elem) {
  const int n  = lane & 15;
  const int kh = (lane >> 4) << 3;
  v16bf b;
#pragma unroll
  for (int i = 0; i < 16; ++i) b[i] = elem(kh + i + (i >= 8 ? 8 : 0), n);
  return b;
}

// D (16x16 f32): VGPR r <-> M = r + 8*(lane>=16), N = lane&15.
template <typename F>
__device__ inline void store_d(int lane, const v8f& d, F&& st) {
  const int n  = lane & 15;
  const int mh = (lane >> 4) << 3;
#pragma unroll
  for (int r = 0; r < 8; ++r) st(mh + r, n, d[r]);
}

// --------------------------------------------------------------------------
// transpose + bf16-convert the four weight matrices: WT[n][k] = W[k][n]
__global__ void k_prep_wT(const float* __restrict__ Wq, const float* __restrict__ Wk,
                          const float* __restrict__ Wv, const float* __restrict__ Wo,
                          __bf16* __restrict__ WqT, __bf16* __restrict__ WkT,
                          __bf16* __restrict__ WvT, __bf16* __restrict__ WoT) {
  int idx = blockIdx.x * blockDim.x + threadIdx.x;
  if (idx >= 4 * DIM * INNER) return;
  int w = idx >> 18;                    // 512*512 = 2^18
  int r = idx & (DIM * INNER - 1);
  int n = r >> 9, k = r & 511;
  const float* W  = (w == 0) ? Wq : (w == 1) ? Wk : (w == 2) ? Wv : Wo;
  __bf16*      WT = (w == 0) ? WqT : (w == 1) ? WkT : (w == 2) ? WvT : WoT;
  WT[(size_t)n * DIM + k] = (__bf16)W[(size_t)k * INNER + n]; // Wk/Wv rows <512
}

// zero the padded key columns of V^T so A@V tail K-steps read zeros
__global__ void k_zero_vpad(__bf16* __restrict__ VbT) {
  int idx = blockIdx.x * blockDim.x + threadIdx.x;
  if (idx >= BATCH * INNER * 16) return;
  int c = idx & 15;
  int d = (idx >> 4) & (INNER - 1);
  int b = idx >> 13;                    // 512*16 = 2^13
  VbT[((size_t)b * INNER + d) * LPAD + LSEQ + c] = (__bf16)0.f;
}

// Q/Kbase/Vbase projection:  (B*L=544, 512) x (512, 512) x3, WMMA bf16.
// V is written transposed: VbT[b][d][key].
__global__ void __launch_bounds__(32) k_qkv(
    const float* __restrict__ x, const __bf16* __restrict__ WqT,
    const __bf16* __restrict__ WkT, const __bf16* __restrict__ WvT,
    __bf16* __restrict__ Qb, __bf16* __restrict__ Kb, __bf16* __restrict__ VbT) {
  const int lane = threadIdx.x;
  const int mt = blockIdx.x;        // 0..33  (544 rows)
  const int w  = blockIdx.y >> 5;   // 0:Wq 1:Wk 2:Wv
  const int nt = blockIdx.y & 31;   // 0..31
  const __bf16* WT = (w == 0) ? WqT : (w == 1) ? WkT : WvT;
  const int nb = nt * 16;
  v8f acc = {};
  for (int kt = 0; kt < DIM / 32; ++kt) {
    const int kb = kt * 32;
    v16bf a  = load_a_f32_vec(x + (size_t)(mt * 16) * DIM + kb, DIM, lane);
    v16bf bm = load_bT_vec(WT + (size_t)nb * DIM + kb, DIM, lane);
    acc = wmma_bf16(a, bm, acc);
  }
  store_d(lane, acc, [&](int m, int n, float v) {
    int row = mt * 16 + m;
    if (w == 0)      Qb[(size_t)row * INNER + nb + n] = (__bf16)v;
    else if (w == 1) Kb[(size_t)row * INNER + nb + n] = (__bf16)v;
    else {
      int bb = row / LSEQ, t = row % LSEQ;
      VbT[((size_t)bb * INNER + nb + n) * LPAD + t] = (__bf16)v;
    }
  });
}

// qWk[b,h,i,p] = sum_d Q[b,i,h*64+d] * Wk[512+p, h*64+d]
__global__ void k_qwk(const __bf16* __restrict__ Qb, const float* __restrict__ Wk,
                      float* __restrict__ qWk) {
  int idx = blockIdx.x * blockDim.x + threadIdx.x;
  if (idx >= BATCH * HEADS * LS * POS_DIM) return;
  int p = idx % POS_DIM;
  int i = (idx / POS_DIM) % LS;
  int h = (idx / (POS_DIM * LS)) % HEADS;
  int b = idx / (POS_DIM * LS * HEADS);
  const __bf16* qrow = Qb + (size_t)(b * LSEQ + i) * INNER + h * DIM_HEAD;
  const float*  wrow = Wk + (size_t)(DIM + p) * INNER + h * DIM_HEAD;
  float acc = 0.f;
  for (int d = 0; d < DIM_HEAD; ++d) acc += (float)qrow[d] * wrow[d];
  qWk[idx] = acc;
}

// base scores: Sc[b,h,i,j] = scale * Q . Kbase over full 272x272, WMMA
__global__ void __launch_bounds__(32) k_scores_base(
    const __bf16* __restrict__ Qb, const __bf16* __restrict__ Kb,
    float* __restrict__ Sc) {
  const int lane = threadIdx.x;
  const int it = blockIdx.x / 17, jt = blockIdx.x % 17;
  const int bh = blockIdx.y, b = bh >> 3, h = bh & 7;
  v8f acc = {};
  for (int kt = 0; kt < DIM_HEAD / 32; ++kt) {
    const int kb = h * DIM_HEAD + kt * 32;
    v16bf a  = load_a_vec(Qb + (size_t)(b * LSEQ + it * 16) * INNER + kb, INNER, lane);
    // K rows are contiguous along d -> K^T B-operand is a straight vector load
    v16bf bm = load_bT_vec(Kb + (size_t)(b * LSEQ + jt * 16) * INNER + kb, INNER, lane);
    acc = wmma_bf16(a, bm, acc);
  }
  store_d(lane, acc, [&](int m, int n, float v) {
    Sc[((size_t)bh * LSEQ + it * 16 + m) * LSEQ + jt * 16 + n] = v * ATTN_SCALE;
  });
}

// pe score contribution, per (b,i):  Sc[h, j] += scale * pe(i,j,:) . qWk[h,:]
// pe tile (16 j x 160 p, bf16) staged in LDS; A-operand = pe (contig p).
__global__ void __launch_bounds__(32) k_scores_pe(
    const float* __restrict__ pos, const float* __restrict__ freqs,
    const float* __restrict__ qWk, float* __restrict__ Sc) {
  __shared__ __bf16 pe[16][PPAD];
  const int lane = threadIdx.x;
  const int b = blockIdx.x >> 8, i = blockIdx.x & 255;
  const float pxi = pos[(size_t)(b * LS + i) * 2];
  const float pyi = pos[(size_t)(b * LS + i) * 2 + 1];
  for (int jt = 0; jt < LS / 16; ++jt) {
    __syncthreads();
    const int jl = lane & 15, half = lane >> 4;
    const int j = jt * 16 + jl;
    float dx = pos[(size_t)(b * LS + j) * 2] - pxi;
    float dy = pos[(size_t)(b * LS + j) * 2 + 1] - pyi;
    float r  = sqrtf(dx * dx + dy * dy + 1e-8f);
    float rc = r / (PHYS_SCALE + r);
    float tn = atan2f(dy, dx) * (1.f / PI_F);
    for (int f = half * 16; f < half * 16 + 16; ++f) {
      float fr = freqs[f] * PI_F, s, c;
      __sincosf(rc * fr, &s, &c);
      pe[jl][f] = (__bf16)s;      pe[jl][32 + f] = (__bf16)c;
      __sincosf(tn * fr, &s, &c);
      pe[jl][65 + f] = (__bf16)s; pe[jl][97 + f] = (__bf16)c;
    }
    if (half == 0) {
      pe[jl][64] = (__bf16)rc; pe[jl][129] = (__bf16)tn;
      for (int p = 130; p < 145; ++p) pe[jl][p] = (__bf16)0.f;
    } else {
      for (int p = 145; p < PPAD; ++p) pe[jl][p] = (__bf16)0.f;
    }
    __syncthreads();
    v8f acc = {};
    for (int kt = 0; kt < PPAD / 32; ++kt) {
      const int kb = kt * 32;
      v16bf a  = load_a_vec(&pe[0][kb], PPAD, lane);   // M=j, K=p (LDS b128)
      v16bf bm = load_b_gen(lane, [&](int k, int n) -> __bf16 {
        int p = kb + k;                                // N=h (guarded tail)
        return (n < HEADS && p < POS_DIM)
                 ? (__bf16)qWk[((size_t)(b * HEADS + n) * LS + i) * POS_DIM + p]
                 : (__bf16)0.f;
      });
      acc = wmma_bf16(a, bm, acc);
    }
    store_d(lane, acc, [&](int m, int n, float v) {   // m=j_local, n=h
      if (n < HEADS)
        Sc[((size_t)(b * HEADS + n) * LSEQ + i) * LSEQ + jt * 16 + m] +=
            v * ATTN_SCALE;
    });
  }
}

// row softmax over 272, writes bf16 probs padded to 288 cols (pad = 0)
__global__ void k_softmax(const float* __restrict__ Sc, __bf16* __restrict__ PB) {
  __shared__ float red[9];
  const int t = threadIdx.x;                 // 0..287 (9 waves)
  const size_t row = blockIdx.x;             // (b*8+h)*272 + i
  float v = (t < LSEQ) ? Sc[row * LSEQ + t] : -3.0e38f;
  float m = v;
  for (int o = 16; o > 0; o >>= 1) m = fmaxf(m, __shfl_xor(m, o, 32));
  if ((t & 31) == 0) red[t >> 5] = m;
  __syncthreads();
  if (t == 0) { float mm = red[0]; for (int w = 1; w < 9; ++w) mm = fmaxf(mm, red[w]); red[0] = mm; }
  __syncthreads();
  m = red[0];
  float e = (t < LSEQ) ? __expf(v - m) : 0.f;
  __syncthreads();
  float s = e;
  for (int o = 16; o > 0; o >>= 1) s += __shfl_xor(s, o, 32);
  if ((t & 31) == 0) red[t >> 5] = s;
  __syncthreads();
  if (t == 0) { float ss = 0.f; for (int w = 0; w < 9; ++w) ss += red[w]; red[0] = ss; }
  __syncthreads();
  PB[row * LPAD + t] = (__bf16)(e / red[0]);
}

// O = P @ Vbase over all 288(pad) keys, WMMA; V^T gives contiguous B loads
__global__ void __launch_bounds__(32) k_out_base(
    const __bf16* __restrict__ PB, const __bf16* __restrict__ VbT,
    float* __restrict__ O) {
  const int lane = threadIdx.x;
  const int it = blockIdx.x >> 2, nt = blockIdx.x & 3;
  const int bh = blockIdx.y, b = bh >> 3, h = bh & 7;
  v8f acc = {};
  for (int kt = 0; kt < LPAD / 32; ++kt) {
    const int kb = kt * 32;
    v16bf a  = load_a_vec(PB + ((size_t)bh * LSEQ + it * 16) * LPAD + kb, LPAD, lane);
    v16bf bm = load_bT_vec(
        VbT + ((size_t)b * INNER + h * DIM_HEAD + nt * 16) * LPAD + kb, LPAD, lane);
    acc = wmma_bf16(a, bm, acc);
  }
  store_d(lane, acc, [&](int m, int n, float v) {
    O[((size_t)bh * LSEQ + it * 16 + m) * DIM_HEAD + nt * 16 + n] = v;
  });
}

// per (b,i):  S[h,p] = sum_{j<256} P[h,j]*pe(i,j,p)  via WMMA; pe stored
// transposed in LDS (peT[p][j]) so the B-operand is a contiguous b128 load.
// Then O[h,d] += S[h,:] @ Wv_pe[:, h*64+d]  (block-diagonal, VALU).
__global__ void __launch_bounds__(32) k_out_pe(
    const float* __restrict__ pos, const float* __restrict__ freqs,
    const __bf16* __restrict__ PB, const float* __restrict__ Wv,
    float* __restrict__ O) {
  __shared__ __bf16 peT[PPAD][32];
  __shared__ float  Sld[16][PPAD];
  const int lane = threadIdx.x;
  const int b = blockIdx.x >> 8, i = blockIdx.x & 255;
  const float pxi = pos[(size_t)(b * LS + i) * 2];
  const float pyi = pos[(size_t)(b * LS + i) * 2 + 1];
  v8f zero = {};
  v8f acc[PPAD / 16];
#pragma unroll
  for (int pt = 0; pt < PPAD / 16; ++pt) acc[pt] = zero;

  for (int jt = 0; jt < LS / 32; ++jt) {
    __syncthreads();
    const int j = jt * 32 + lane;
    float dx = pos[(size_t)(b * LS + j) * 2] - pxi;
    float dy = pos[(size_t)(b * LS + j) * 2 + 1] - pyi;
    float r  = sqrtf(dx * dx + dy * dy + 1e-8f);
    float rc = r / (PHYS_SCALE + r);
    float tn = atan2f(dy, dx) * (1.f / PI_F);
    for (int f = 0; f < NFREQ; ++f) {
      float fr = freqs[f] * PI_F, s, c;
      __sincosf(rc * fr, &s, &c);
      peT[f][lane] = (__bf16)s;      peT[32 + f][lane] = (__bf16)c;
      __sincosf(tn * fr, &s, &c);
      peT[65 + f][lane] = (__bf16)s; peT[97 + f][lane] = (__bf16)c;
    }
    peT[64][lane] = (__bf16)rc; peT[129][lane] = (__bf16)tn;
    for (int p = POS_DIM; p < PPAD; ++p) peT[p][lane] = (__bf16)0.f;
    __syncthreads();
    // A: M=h (8 real rows), K=j -- predicated vector load from PB
    const int m = lane & 15, kh = (lane >> 4) << 3;
    v16bf a;
#pragma unroll
    for (int q = 0; q < 16; ++q) a[q] = (__bf16)0.f;
    if (m < HEADS) {
      const __bf16* rp = PB + ((size_t)(b * HEADS + m) * LSEQ + i) * LPAD + jt * 32;
      v8bf lo = *(const v8bf*)(rp + kh);
      v8bf hi = *(const v8bf*)(rp + 16 + kh);
      a = __builtin_shufflevector(lo, hi, 0, 1, 2, 3, 4, 5, 6, 7,
                                  8, 9, 10, 11, 12, 13, 14, 15);
    }
#pragma unroll
    for (int pt = 0; pt < PPAD / 16; ++pt) {
      v16bf bm = load_bT_vec(&peT[pt * 16][0], 32, lane);      // K=j, N=p
      acc[pt] = wmma_bf16(a, bm, acc[pt]);
    }
  }
  __syncthreads();
#pragma unroll
  for (int pt = 0; pt < PPAD / 16; ++pt)
    store_d(lane, acc[pt], [&](int m2, int n2, float v) { Sld[m2][pt * 16 + n2] = v; });
  __syncthreads();
  for (int o = lane; o < HEADS * DIM_HEAD; o += 32) {
    int h = o >> 6, d = o & 63;
    float s = 0.f;
    for (int p = 0; p < POS_DIM; ++p)
      s += Sld[h][p] * Wv[(size_t)(DIM + p) * INNER + h * DIM_HEAD + d];
    O[((size_t)(b * HEADS + h) * LSEQ + i) * DIM_HEAD + d] += s;
  }
}

// out = concat_heads(O) @ Wo + bo, WMMA (Wo pre-transposed to bf16)
__global__ void __launch_bounds__(32) k_out_proj(
    const float* __restrict__ O, const __bf16* __restrict__ WoT,
    const float* __restrict__ bo, float* __restrict__ out) {
  const int lane = threadIdx.x;
  const int mt = blockIdx.x, nt = blockIdx.y;
  v8f acc = {};
  for (int kt = 0; kt < INNER / 32; ++kt) {
    const int kb = kt * 32;
    // A: gather the head-concatenated row of O with two aligned 32B chunks
    const int m = lane & 15, kh = (lane >> 4) << 3;
    int row = mt * 16 + m, bb = row / LSEQ, t = row % LSEQ;
    int k0 = kb + kh, k1 = kb + 16 + kh;   // chunks never cross a head boundary
    const float* p0 = O + ((size_t)(bb * HEADS + (k0 >> 6)) * LSEQ + t) * DIM_HEAD + (k0 & 63);
    const float* p1 = O + ((size_t)(bb * HEADS + (k1 >> 6)) * LSEQ + t) * DIM_HEAD + (k1 & 63);
    v4f a0 = *(const v4f*)(p0); v4f a1 = *(const v4f*)(p0 + 4);
    v4f a2 = *(const v4f*)(p1); v4f a3 = *(const v4f*)(p1 + 4);
    v16bf a;
#pragma unroll
    for (int q = 0; q < 4; ++q) {
      a[q]      = (__bf16)a0[q];
      a[4 + q]  = (__bf16)a1[q];
      a[8 + q]  = (__bf16)a2[q];
      a[12 + q] = (__bf16)a3[q];
    }
    v16bf bm = load_bT_vec(WoT + (size_t)(nt * 16) * DIM + kb, DIM, lane);
    acc = wmma_bf16(a, bm, acc);
  }
  store_d(lane, acc, [&](int m, int n, float v) {
    out[(size_t)(mt * 16 + m) * DIM + nt * 16 + n] = v + bo[nt * 16 + n];
  });
}

// ---------------------------------------------------------------------------
extern "C" void kernel_launch(void* const* d_in, const int* in_sizes, int n_in,
                              void* d_out, int out_size, void* d_ws, size_t ws_size,
                              hipStream_t stream) {
  (void)in_sizes; (void)n_in; (void)out_size; (void)ws_size;
  const float* x     = (const float*)d_in[0];
  const float* pos   = (const float*)d_in[1];
  const float* Wq    = (const float*)d_in[2];
  const float* Wk    = (const float*)d_in[3];
  const float* Wv    = (const float*)d_in[4];
  const float* Wo    = (const float*)d_in[5];
  const float* bo    = (const float*)d_in[6];
  const float* freqs = (const float*)d_in[7];
  float* out = (float*)d_out;

  char* ws = (char*)d_ws;
  size_t off = 0;
  auto alloc = [&](size_t bytes) -> void* {
    void* p = ws + off;
    off += (bytes + 255) & ~(size_t)255;
    return p;
  };
  __bf16* WqT = (__bf16*)alloc((size_t)DIM * INNER * 2);
  __bf16* WkT = (__bf16*)alloc((size_t)DIM * INNER * 2);
  __bf16* WvT = (__bf16*)alloc((size_t)DIM * INNER * 2);
  __bf16* WoT = (__bf16*)alloc((size_t)DIM * INNER * 2);
  __bf16* Qb  = (__bf16*)alloc((size_t)BATCH * LSEQ * INNER * 2);
  __bf16* Kb  = (__bf16*)alloc((size_t)BATCH * LSEQ * INNER * 2);
  __bf16* VbT = (__bf16*)alloc((size_t)BATCH * INNER * LPAD * 2);
  float*  qWk = (float*) alloc((size_t)BATCH * HEADS * LS * POS_DIM * 4);
  float*  Sc  = (float*) alloc((size_t)BATCH * HEADS * LSEQ * LSEQ * 4);
  __bf16* PB  = (__bf16*)alloc((size_t)BATCH * HEADS * LSEQ * LPAD * 2);
  float*  O   = (float*) alloc((size_t)BATCH * HEADS * LSEQ * DIM_HEAD * 4);

  k_prep_wT   <<<dim3((4 * DIM * INNER + 255) / 256), 256, 0, stream>>>(
                  Wq, Wk, Wv, Wo, WqT, WkT, WvT, WoT);
  k_zero_vpad <<<dim3((BATCH * INNER * 16 + 255) / 256), 256, 0, stream>>>(VbT);
  k_qkv       <<<dim3(34, 96), 32, 0, stream>>>(x, WqT, WkT, WvT, Qb, Kb, VbT);
  k_qwk       <<<dim3((BATCH * HEADS * LS * POS_DIM + 255) / 256), 256, 0, stream>>>(Qb, Wk, qWk);
  k_scores_base<<<dim3(17 * 17, 16), 32, 0, stream>>>(Qb, Kb, Sc);
  k_scores_pe <<<dim3(BATCH * LS), 32, 0, stream>>>(pos, freqs, qWk, Sc);
  k_softmax   <<<dim3(BATCH * HEADS * LSEQ), 288, 0, stream>>>(Sc, PB);
  k_out_base  <<<dim3(17 * 4, 16), 32, 0, stream>>>(PB, VbT, O);
  k_out_pe    <<<dim3(BATCH * LS), 32, 0, stream>>>(pos, freqs, PB, Wv, O);
  k_out_proj  <<<dim3(34, 32), 32, 0, stream>>>(O, WoT, bo, out);
}
